// MambaLayer_53051436040366
// MI455X (gfx1250) — matrix operand order repaired and checked by reading, not compiled, not run on hardware
//
#include <hip/hip_runtime.h>
#include <hip/hip_bf16.h>

// ---------------------------------------------------------------------------
// Types for CDNA5 WMMA
// ---------------------------------------------------------------------------
typedef __attribute__((ext_vector_type(16))) _Float16 v16h;
typedef __attribute__((ext_vector_type(8)))  _Float16 v8h;
typedef __attribute__((ext_vector_type(8)))  float    v8f;

union AFrag { v16h v; v8h h[2]; };

#define LSEQ   2048
#define BATCH  2
#define DIMC   768
#define DINNER 1536
#define BLROWS 4096          // BATCH*LSEQ
#define NXPP   96            // DT_RANK + 2*D_STATE (80) padded to 96 for N%32==0
#define KDT    64            // DT_RANK (48) padded to 64

// Epilogue modes for the fused GEMM store
#define EP_NONE          0   // C[i] = acc
#define EP_BIAS_GELU_F16 1   // C16[i] = (f16) gelu(acc + bias[col])
#define EP_BIAS_RES      2   // C[i] = acc + bias[col] + res[i]
#define EP_RES           3   // C[i] = acc + res[i]

// ---------------------------------------------------------------------------
// Register-blocked WMMA GEMM: C[M,N] (f32) = A[M,K] (f16 RM) * W[N,K]^T (f16)
// One wave -> 64x32 macro-tile: 4 M-tiles x 2 N-tiles, 8 accumulators.
// All 6 fragment loads of a K-step are issued before the 8 WMMAs so the
// scheduler can drain the load clause with partial s_wait_loadcnt while
// earlier WMMAs execute.
// Requires M%64==0, N%32==0, K%32==0, ((M/64)*(N/32))%8==0.
// ---------------------------------------------------------------------------
__global__ __launch_bounds__(256) void wmma_gemm_blk(
    const _Float16* __restrict__ A, const _Float16* __restrict__ W,
    float* __restrict__ C, _Float16* __restrict__ C16,
    const float* __restrict__ bias, const float* __restrict__ res,
    int M, int N, int K, int mode) {
  const int lane = threadIdx.x & 31;
  const int wave = threadIdx.x >> 5;
  const int nBlkN = N >> 5;                       // N/32
  const int tile = blockIdx.x * 8 + wave;
  if (tile >= (M >> 6) * nBlkN) return;
  const int bm = tile / nBlkN;
  const int bn = tile % nBlkN;
  const int m0 = bm << 6;                         // 64-row macro tile
  const int n0 = bn << 5;                         // 32-col macro tile
  const int nl = lane & 15;
  const int hb = lane >> 4;

  size_t arow[4], brow[2];
#pragma unroll
  for (int mi = 0; mi < 4; ++mi) arow[mi] = (size_t)(m0 + mi * 16 + nl) * K;
#pragma unroll
  for (int ni = 0; ni < 2; ++ni) brow[ni] = (size_t)(n0 + ni * 16 + nl) * K;

  v8f acc[8];
#pragma unroll
  for (int i = 0; i < 8; ++i) acc[i] = (v8f){};

  for (int k0 = 0; k0 < K; k0 += 32) {
    // ---- issue ALL fragment loads for this K-step first ----
    // B 32x16 f16, this code's contiguous k order (same as A): lanes 0-15 K=k0..k0+15, lanes 16-31 K=k0+16..k0+31
    v16h b0 = *(const v16h*)(W + brow[0] + k0 + hb * 16);
    v16h b1 = *(const v16h*)(W + brow[1] + k0 + hb * 16);
    // A 16x32 f16, this code's contiguous k order (same as B): VGPR0-3 => K=k0+hb*16+0..7, VGPR4-7 => +8
    AFrag a[4];
#pragma unroll
    for (int mi = 0; mi < 4; ++mi) {
      a[mi].h[0] = *(const v8h*)(A + arow[mi] + k0 + hb * 16);
      a[mi].h[1] = *(const v8h*)(A + arow[mi] + k0 + hb * 16 + 8);
    }
    if (k0 + 32 < K) {
      __builtin_prefetch(W + brow[0] + k0 + 32, 0, 1);   // global_prefetch_b8
      __builtin_prefetch(A + arow[0] + k0 + 32, 0, 1);
    }
    // ---- then the 8 WMMAs (B frags reused 4x, A frags 2x) ----
#pragma unroll
    for (int mi = 0; mi < 4; ++mi) {
      acc[mi * 2 + 0] = __builtin_amdgcn_wmma_f32_16x16x32_f16(
          false, a[mi].v, false, b0, (short)0, acc[mi * 2 + 0], false, false);
      acc[mi * 2 + 1] = __builtin_amdgcn_wmma_f32_16x16x32_f16(
          false, a[mi].v, false, b1, (short)0, acc[mi * 2 + 1], false, false);
    }
  }

  // C/D f32 16x16 layout: VGPR r -> row hb*8 + r, lane nl -> column nl
#pragma unroll
  for (int mi = 0; mi < 4; ++mi) {
#pragma unroll
    for (int ni = 0; ni < 2; ++ni) {
      const int row0 = m0 + mi * 16 + hb * 8;
      const int col  = n0 + ni * 16 + nl;
      const v8f a8 = acc[mi * 2 + ni];
#pragma unroll
      for (int r = 0; r < 8; ++r) {
        const size_t idx = (size_t)(row0 + r) * N + col;
        float v = a8[r];
        if (mode == EP_BIAS_GELU_F16) {
          v += bias[col];
          v = 0.5f * v * (1.f + erff(v * 0.70710678118f));
          C16[idx] = (_Float16)v;
        } else if (mode == EP_BIAS_RES) {
          C[idx] = v + bias[col] + res[idx];
        } else if (mode == EP_RES) {
          C[idx] = v + res[idx];
        } else {
          C[idx] = v;
        }
      }
    }
  }
}

// ---------------------------------------------------------------------------
// Helpers: conversions / fills
// ---------------------------------------------------------------------------
__global__ void convert_f32_f16(const float* __restrict__ in,
                                _Float16* __restrict__ out, size_t n) {
  size_t i = (size_t)blockIdx.x * blockDim.x + threadIdx.x;
  if (i < n) out[i] = (_Float16)in[i];
}

// zero-padded convert: out[r, 0..cout) = (c < cin) ? in[r*stride+c] : 0
__global__ void convert_pad_f16(const float* __restrict__ in,
                                _Float16* __restrict__ out,
                                int rows, int cin, int stride_in, int cout) {
  size_t i = (size_t)blockIdx.x * blockDim.x + threadIdx.x;
  size_t n = (size_t)rows * cout;
  if (i >= n) return;
  int c = (int)(i % cout);
  int r = (int)(i / cout);
  out[i] = (c < cin) ? (_Float16)in[(size_t)r * stride_in + c] : (_Float16)0.f;
}

__global__ void fill_zero_f32(float* __restrict__ p, size_t n) {
  size_t i = (size_t)blockIdx.x * blockDim.x + threadIdx.x;
  if (i < n) p[i] = 0.f;
}

__global__ void fill_zero_f16(_Float16* __restrict__ p, size_t n) {
  size_t i = (size_t)blockIdx.x * blockDim.x + threadIdx.x;
  if (i < n) p[i] = (_Float16)0.f;
}

// ---------------------------------------------------------------------------
// LayerNorm over last dim (ncols), block per row, f16 output for WMMA input.
// ---------------------------------------------------------------------------
__global__ __launch_bounds__(256) void layernorm_to_f16(
    const float* __restrict__ x, const float* __restrict__ g,
    const float* __restrict__ bb, _Float16* __restrict__ out, int ncols) {
  __shared__ float red0[8], red1[8];
  const int row = blockIdx.x;
  const float* xr = x + (size_t)row * ncols;
  float s = 0.f, s2 = 0.f;
  for (int c = threadIdx.x; c < ncols; c += blockDim.x) {
    float v = xr[c]; s += v; s2 += v * v;
  }
  for (int o = 16; o > 0; o >>= 1) {
    s += __shfl_down(s, o, 32); s2 += __shfl_down(s2, o, 32);
  }
  const int lane = threadIdx.x & 31, wave = threadIdx.x >> 5;
  if (lane == 0) { red0[wave] = s; red1[wave] = s2; }
  __syncthreads();
  if (wave == 0) {
    s  = (lane < 8) ? red0[lane] : 0.f;
    s2 = (lane < 8) ? red1[lane] : 0.f;
    for (int o = 4; o > 0; o >>= 1) {
      s += __shfl_down(s, o, 32); s2 += __shfl_down(s2, o, 32);
    }
    if (lane == 0) { red0[0] = s; red1[0] = s2; }
  }
  __syncthreads();
  const float mu  = red0[0] / ncols;
  const float var = red1[0] / ncols - mu * mu;
  const float inv = rsqrtf(var + 1e-5f);
  for (int c = threadIdx.x; c < ncols; c += blockDim.x) {
    float v = (xr[c] - mu) * inv * g[c] + bb[c];
    out[(size_t)row * ncols + c] = (_Float16)v;
  }
}

// ---------------------------------------------------------------------------
// Depthwise causal conv1d (K=4) + SiLU, with optional time reversal.
// ---------------------------------------------------------------------------
__global__ __launch_bounds__(256) void conv_silu(
    const float* __restrict__ xz, const float* __restrict__ cw,
    const float* __restrict__ cb, float* __restrict__ xc,
    _Float16* __restrict__ xch, int rev) {
  size_t i = (size_t)blockIdx.x * blockDim.x + threadIdx.x;
  size_t total = (size_t)BATCH * LSEQ * DINNER;
  if (i >= total) return;
  const int d = (int)(i % DINNER);
  const int s = (int)((i / DINNER) % LSEQ);
  const int b = (int)(i / ((size_t)LSEQ * DINNER));
  float a = cb[d];
#pragma unroll
  for (int k = 0; k < 4; ++k) {
    int sp = s - 3 + k;
    if (sp >= 0) {
      int lo = rev ? (LSEQ - 1 - sp) : sp;
      a += cw[d * 4 + k] * xz[((size_t)b * LSEQ + lo) * (2 * DINNER) + d];
    }
  }
  float v = a / (1.f + expf(-a));   // silu
  xc[i]  = v;
  xch[i] = (_Float16)v;
}

// ---------------------------------------------------------------------------
// Selective scan (sequential over L), one thread per (b, d_inner) channel.
// Per timestep the shared B/C (32 floats) are staged via LDS.
// ---------------------------------------------------------------------------
__global__ __launch_bounds__(256) void scan_acc(
    const float* __restrict__ xc, const float* __restrict__ xdbl,
    const float* __restrict__ dtlin, const float* __restrict__ dt_b,
    const float* __restrict__ A_log, const float* __restrict__ Dskip,
    const float* __restrict__ xz, float* __restrict__ acc, int rev) {
  const int grp = DINNER / 256;                     // 6
  const int b = blockIdx.x / grp;
  const int d = (blockIdx.x % grp) * 256 + threadIdx.x;
  float Ar[16], h[16];
#pragma unroll
  for (int n = 0; n < 16; ++n) {
    Ar[n] = -expf(A_log[d * 16 + n]);
    h[n] = 0.f;
  }
  const float dtb = dt_b[d];
  const float Dv  = Dskip[d];
  __shared__ float sBC[32];
  for (int s = 0; s < LSEQ; ++s) {
    __syncthreads();
    if (threadIdx.x < 32)
      sBC[threadIdx.x] = xdbl[((size_t)b * LSEQ + s) * NXPP + 48 + threadIdx.x];
    __syncthreads();
    const size_t ro = (size_t)b * LSEQ + s;
    const float u  = xc[ro * DINNER + d];
    const float dl = dtlin[ro * DINNER + d] + dtb;
    const float dt = (dl > 20.f) ? dl : log1pf(expf(dl));   // softplus
    const float du = dt * u;
    float y = 0.f;
#pragma unroll
    for (int n = 0; n < 16; ++n) {
      h[n] = expf(dt * Ar[n]) * h[n] + du * sBC[n];
      y += h[n] * sBC[16 + n];
    }
    y += u * Dv;
    const int lo = rev ? (LSEQ - 1 - s) : s;
    const size_t oo = (size_t)b * LSEQ + lo;
    const float zv = xz[oo * (2 * DINNER) + DINNER + d];
    const float gate = zv / (1.f + expf(-zv));              // silu(z)
    acc[oo * DINNER + d] += y * gate;
  }
}

// ---------------------------------------------------------------------------
// Host orchestration
// ---------------------------------------------------------------------------
static inline int blocks_for(size_t n, int bs) { return (int)((n + bs - 1) / bs); }

extern "C" void kernel_launch(void* const* d_in, const int* in_sizes, int n_in,
                              void* d_out, int out_size, void* d_ws, size_t ws_size,
                              hipStream_t stream) {
  const float* x         = (const float*)d_in[0];
  const float* g1        = (const float*)d_in[1];
  const float* b1        = (const float*)d_in[2];
  const float* in_proj_w = (const float*)d_in[3];
  const float* conv_w    = (const float*)d_in[4];
  const float* conv_b    = (const float*)d_in[5];
  const float* x_proj_w  = (const float*)d_in[6];
  const float* dt_proj_w = (const float*)d_in[7];
  const float* dt_proj_b = (const float*)d_in[8];
  const float* A_log     = (const float*)d_in[9];
  const float* A_b_log   = (const float*)d_in[10];
  const float* Dskip     = (const float*)d_in[11];
  const float* out_proj_w= (const float*)d_in[12];
  const float* g2        = (const float*)d_in[13];
  const float* b2        = (const float*)d_in[14];
  const float* fc1_w     = (const float*)d_in[15];
  const float* fc1_b     = (const float*)d_in[16];
  const float* fc2_w     = (const float*)d_in[17];
  const float* fc2_b     = (const float*)d_in[18];
  float* out = (float*)d_out;

  // ---- workspace carve-up (bump allocator, 256B aligned) ----
  char* ws = (char*)d_ws;
  size_t off = 0;
  auto take = [&](size_t bytes) -> char* {
    char* p = ws + off;
    off += (bytes + 255) & ~(size_t)255;
    return p;
  };
  _Float16* w_in_h  = (_Float16*)take((size_t)2 * DINNER * DIMC * 2);     // 3072x768
  _Float16* w_xp_h  = (_Float16*)take((size_t)NXPP * DINNER * 2);         // 96x1536 (rows 80..95 zero)
  _Float16* w_dt_h  = (_Float16*)take((size_t)DINNER * KDT * 2);          // 1536x64 (K padded)
  _Float16* w_out_h = (_Float16*)take((size_t)DIMC * DINNER * 2);         // 768x1536
  _Float16* w_fc1_h = (_Float16*)take((size_t)2 * DIMC * DIMC * 2);       // 1536x768
  _Float16* w_fc2_h = (_Float16*)take((size_t)DIMC * 2 * DIMC * 2);       // 768x1536
  _Float16* ln_h    = (_Float16*)take((size_t)BLROWS * DIMC * 2);         // ln1 & ln2 (reused)
  float*    xz      = (float*)take((size_t)BLROWS * 2 * DINNER * 4);      // 4096x3072
  float*    accbuf  = (float*)take((size_t)BLROWS * DINNER * 4);          // gated out_f+out_b
  float*    xc      = (float*)take((size_t)BLROWS * DINNER * 4);
  _Float16* act_h   = (_Float16*)take((size_t)BLROWS * DINNER * 2);       // xc_f16 & ffn act (reused)
  float*    xdbl    = (float*)take((size_t)BLROWS * NXPP * 4);            // 4096x96
  _Float16* dtr_h   = (_Float16*)take((size_t)BLROWS * KDT * 2);
  float*    big1536 = (float*)take((size_t)BLROWS * DINNER * 4);          // dtlin (reused)
  _Float16* acc_h   = (_Float16*)take((size_t)BLROWS * DINNER * 2);
  float*    hbuf    = (float*)take((size_t)BLROWS * DIMC * 4);
  (void)ws_size; (void)in_sizes; (void)n_in; (void)out_size;

  auto gemm = [&](const _Float16* A, const _Float16* W, float* C, _Float16* C16,
                  const float* bias, const float* res, int M, int N, int K,
                  int mode) {
    int tiles = (M / 64) * (N / 32);
    wmma_gemm_blk<<<tiles / 8, 256, 0, stream>>>(A, W, C, C16, bias, res,
                                                 M, N, K, mode);
  };

  // ---- 1) weight conversions to f16 (once per call) ----
  {
    size_t n;
    n = (size_t)2 * DINNER * DIMC;
    convert_f32_f16<<<blocks_for(n, 256), 256, 0, stream>>>(in_proj_w, w_in_h, n);
    // x_proj: first 80 rows converted, rows 80..95 zeroed (N padded to 96)
    n = (size_t)80 * DINNER;
    convert_f32_f16<<<blocks_for(n, 256), 256, 0, stream>>>(x_proj_w, w_xp_h, n);
    {
      size_t ntail = (size_t)(NXPP - 80) * DINNER;
      fill_zero_f16<<<blocks_for(ntail, 256), 256, 0, stream>>>(
          w_xp_h + (size_t)80 * DINNER, ntail);
    }
    n = (size_t)DINNER * KDT;   // pad K 48 -> 64
    convert_pad_f16<<<blocks_for(n, 256), 256, 0, stream>>>(dt_proj_w, w_dt_h,
                                                            DINNER, 48, 48, KDT);
    n = (size_t)DIMC * DINNER;
    convert_f32_f16<<<blocks_for(n, 256), 256, 0, stream>>>(out_proj_w, w_out_h, n);
    n = (size_t)DINNER * DIMC;
    convert_f32_f16<<<blocks_for(n, 256), 256, 0, stream>>>(fc1_w, w_fc1_h, n);
    n = (size_t)DIMC * DINNER;
    convert_f32_f16<<<blocks_for(n, 256), 256, 0, stream>>>(fc2_w, w_fc2_h, n);
  }

  // ---- 2) ln1 -> f16, 3) in_proj GEMM ----
  layernorm_to_f16<<<BLROWS, 256, 0, stream>>>(x, g1, b1, ln_h, DIMC);
  gemm(ln_h, w_in_h, xz, nullptr, nullptr, nullptr,
       BLROWS, 2 * DINNER, DIMC, EP_NONE);

  // ---- 4) zero gated accumulator ----
  {
    size_t n = (size_t)BLROWS * DINNER;
    fill_zero_f32<<<blocks_for(n, 256), 256, 0, stream>>>(accbuf, n);
  }

  // ---- 5) per-direction: conv+silu, x_proj, dt_proj, scan ----
  for (int dir = 0; dir < 2; ++dir) {
    const float* Adir = dir ? A_b_log : A_log;
    size_t nconv = (size_t)BATCH * LSEQ * DINNER;
    conv_silu<<<blocks_for(nconv, 256), 256, 0, stream>>>(xz, conv_w, conv_b,
                                                          xc, act_h, dir);
    gemm(act_h, w_xp_h, xdbl, nullptr, nullptr, nullptr,
         BLROWS, NXPP, DINNER, EP_NONE);
    {
      size_t n = (size_t)BLROWS * KDT;  // dt_r slice -> f16, K padded 48->64
      convert_pad_f16<<<blocks_for(n, 256), 256, 0, stream>>>(xdbl, dtr_h,
                                                              BLROWS, 48, NXPP, KDT);
    }
    gemm(dtr_h, w_dt_h, big1536, nullptr, nullptr, nullptr,
         BLROWS, DINNER, KDT, EP_NONE);
    scan_acc<<<BATCH * (DINNER / 256), 256, 0, stream>>>(
        xc, xdbl, big1536, dt_proj_b, Adir, Dskip, xz, accbuf, dir);
  }

  // ---- 6-7) out_proj (+ residual x fused) -> hbuf ----
  {
    size_t n = (size_t)BLROWS * DINNER;
    convert_f32_f16<<<blocks_for(n, 256), 256, 0, stream>>>(accbuf, acc_h, n);
  }
  gemm(acc_h, w_out_h, hbuf, nullptr, nullptr, x,
       BLROWS, DIMC, DINNER, EP_RES);

  // ---- 8-10) FFN: ln2 -> fc1(+bias+gelu->f16) -> fc2(+bias+residual->out) ----
  layernorm_to_f16<<<BLROWS, 256, 0, stream>>>(hbuf, g2, b2, ln_h, DIMC);
  gemm(ln_h, w_fc1_h, nullptr, act_h, fc1_b, nullptr,
       BLROWS, DINNER, DIMC, EP_BIAS_GELU_F16);
  gemm(act_h, w_fc2_h, out, nullptr, fc2_b, hbuf,
       BLROWS, DIMC, DINNER, EP_BIAS_RES);
}
